// ObjAttention_37598143709781
// MI455X (gfx1250) — matrix operand order repaired and checked by reading, not compiled, not run on hardware
//
#include <hip/hip_runtime.h>
#include <hip/hip_bf16.h>
#include <stdint.h>

// ---------------------------------------------------------------------------
// Problem constants (from reference): B=2, N=1024, L=4096, C=1024, H=16, d=64
// ---------------------------------------------------------------------------
#define Bb   2
#define Ndim 1024
#define Ldim 4096
#define Cdim 1024
#define Hh   16
#define Dh   64
#define Mtot (Ndim + Ldim)   // 5120 keys per (b,h)

typedef __attribute__((ext_vector_type(16))) __bf16 v16bf;
typedef __attribute__((ext_vector_type(8)))  float  v8f;

union Frag { v16bf v; uint4 q[2]; };   // 32 bytes = 8 VGPRs, wave32 16x16 frag

__device__ __forceinline__ unsigned short f2bf(float f) {
    unsigned u = __float_as_uint(f);
    u += 0x7FFFu + ((u >> 16) & 1u);   // round-to-nearest-even
    return (unsigned short)(u >> 16);
}

// ---------------------------------------------------------------------------
// CDNA5 async global->LDS copy (ASYNCcnt-tracked); falls back to sync copy.
// Probe-derived signature: param0 = AS(1) pointer to vector_size(16) int.
// ---------------------------------------------------------------------------
#if defined(__has_builtin)
#if __has_builtin(__builtin_amdgcn_global_load_async_to_lds_b128)
#define HAVE_ASYNC_LDS 1
#endif
#endif

typedef int v4i_vs __attribute__((vector_size(16)));
typedef __attribute__((address_space(1))) v4i_vs* g_b128_ptr;
typedef __attribute__((address_space(3))) v4i_vs* l_b128_ptr;

__device__ __forceinline__ void copy16_g2l(const unsigned short* gsrc,
                                           unsigned short* ldst) {
#ifdef HAVE_ASYNC_LDS
    __builtin_amdgcn_global_load_async_to_lds_b128(
        (g_b128_ptr)gsrc, (l_b128_ptr)ldst, 0, 0);
#else
    *(uint4*)ldst = *(const uint4*)gsrc;
#endif
}

__device__ __forceinline__ void async_wait0() {
#ifdef HAVE_ASYNC_LDS
#if __has_builtin(__builtin_amdgcn_s_wait_asynccnt)
    __builtin_amdgcn_s_wait_asynccnt(0);
#else
    asm volatile("s_wait_asynccnt 0" ::: "memory");
#endif
#endif
}

// ---------------------------------------------------------------------------
// fp32 -> bf16 conversion
// ---------------------------------------------------------------------------
__global__ void cvt_f32_bf16(const float* __restrict__ src,
                             unsigned short* __restrict__ dst, int n) {
    int i = blockIdx.x * blockDim.x + threadIdx.x;
    int stride = gridDim.x * blockDim.x;
    for (; i < n; i += stride) dst[i] = f2bf(src[i]);
}

// ---------------------------------------------------------------------------
// Generic bf16 WMMA GEMM: out = A[rows x Kdim] @ Bm[Kdim x Ncols]
// 128 threads = 4 waves; block tile 64(M) x 64(N); K staged in steps of 32.
// A tile staged via async global->LDS; B staged transposed (sync, needs VGPR
// round-trip). Every WMMA fragment is two aligned ds_load_b128 per lane.
// mode 0: scatter Q  -> outK head-major                    (Ncols = C)
// mode 1: scatter KV -> outK / outV head-major, row moff+t (Ncols = 2C)
// mode 2: f32 out + bias (final projection)
// ---------------------------------------------------------------------------
__global__ __launch_bounds__(128)
void wmma_gemm_bf16(const unsigned short* __restrict__ A,
                    const unsigned short* __restrict__ Bm,
                    int Kd, int Ncols, int mode, int Tper, int moff,
                    unsigned short* __restrict__ outK,
                    unsigned short* __restrict__ outV,
                    float* __restrict__ outF,
                    const float* __restrict__ bias) {
    __shared__ alignas(16) unsigned short Asl[64 * 32];   // [m][k]
    __shared__ alignas(16) unsigned short Btl[64 * 32];   // [n][k] (B^T)

    const int tid  = threadIdx.x;
    const int lane = tid & 31;
    const int w    = tid >> 5;        // wave id 0..3 -> 16-row strip
    const int g    = lane >> 4;       // half-wave group
    const int l15  = lane & 15;

    const int rowBase = blockIdx.y * 64;
    const int colBase = blockIdx.x * 64;

    v8f zero = {0.f,0.f,0.f,0.f,0.f,0.f,0.f,0.f};
    v8f acc[4];
#pragma unroll
    for (int i = 0; i < 4; ++i) acc[i] = zero;

    const int nK = Kd >> 5;
    for (int it = 0; it < nK; ++it) {
        const int k0 = it << 5;
        // ---- stage A: 64x32 bf16 via async b128 copies (256 x 16B) ----
#pragma unroll
        for (int rep = 0; rep < 2; ++rep) {
            int i = tid + (rep << 7);
            int r = i >> 2, c = i & 3;
            copy16_g2l(A + (size_t)(rowBase + r) * Kd + k0 + (c << 3),
                       Asl + r * 32 + (c << 3));
        }
        // ---- stage B^T: Btl[n][k] (transpose through VGPRs) ----
#pragma unroll
        for (int rep = 0; rep < 2; ++rep) {
            int i  = tid + (rep << 7);
            int kk = i >> 3, c8 = i & 7;
            uint4 d = *(const uint4*)(Bm + (size_t)(k0 + kk) * Ncols + colBase + (c8 << 3));
            const unsigned short* e = (const unsigned short*)&d;
#pragma unroll
            for (int t2 = 0; t2 < 8; ++t2)
                Btl[((c8 << 3) + t2) * 32 + kk] = e[t2];
        }
        async_wait0();
        __syncthreads();

        // A fragment: rows [16w,16w+16), k = 16g..16g+15 contiguous
        Frag af;
        {
            const unsigned short* ar = Asl + (w * 16 + l15) * 32 + (g << 4);
            af.q[0] = *(const uint4*)ar;
            af.q[1] = *(const uint4*)(ar + 8);
        }
#pragma unroll
        for (int nt = 0; nt < 4; ++nt) {
            // B fragment: n = nt*16+l15, k = 16g..16g+15 contiguous in B^T row
            Frag bfr;
            const unsigned short* br = Btl + (nt * 16 + l15) * 32 + (g << 4);
            bfr.q[0] = *(const uint4*)br;
            bfr.q[1] = *(const uint4*)(br + 8);
            acc[nt] = __builtin_amdgcn_wmma_f32_16x16x32_bf16(
                false, af.v, false, bfr.v, (short)0, acc[nt], false, false);
        }
        __syncthreads();
    }

    // ---- epilogue: C/D layout: row = v + 8g, col = nt*16 + l15 ----
#pragma unroll
    for (int nt = 0; nt < 4; ++nt) {
#pragma unroll
        for (int v = 0; v < 8; ++v) {
            int m = rowBase + w * 16 + v + (g << 3);
            int c = colBase + nt * 16 + l15;
            float val = acc[nt][v];
            if (mode == 2) {
                outF[(size_t)m * Ncols + c] = val + bias[c];
            } else if (mode == 0) {
                int b = m / Tper, t = m % Tper;
                int h = c >> 6, d = c & 63;
                outK[(((size_t)(b * Hh + h) * Ndim) + t) * Dh + d] = f2bf(val);
            } else {
                int b = m / Tper, t = m % Tper;
                int mm = moff + t;
                if (c < Cdim) {
                    int h = c >> 6, d = c & 63;
                    outK[(((size_t)(b * Hh + h) * Mtot) + mm) * Dh + d] = f2bf(val);
                } else {
                    int c2 = c - Cdim;
                    int h = c2 >> 6, d = c2 & 63;
                    outV[(((size_t)(b * Hh + h) * Mtot) + mm) * Dh + d] = f2bf(val);
                }
            }
        }
    }
}

// ---------------------------------------------------------------------------
// Fused flash-attention: one block per (b, h, 64-query tile); 4 waves, each
// wave owns 16 queries. K tile staged via async global->LDS (row-major K is
// already the B^T layout for Q.K^T); V^T staged through VGPRs (transpose).
// Online softmax in f32; P round-trips LDS as bf16 into A-fragment layout.
// ---------------------------------------------------------------------------
__global__ __launch_bounds__(128)
void flash_attn_bf16(const unsigned short* __restrict__ Q,
                     const unsigned short* __restrict__ K,
                     const unsigned short* __restrict__ V,
                     unsigned short* __restrict__ O) {
    __shared__ alignas(16) unsigned short Ksl[64 * 64];       // [key][d] == B^T for QK^T
    __shared__ alignas(16) unsigned short Vtl[64 * 80];       // [d][key] (pad 80 keeps b128 align)
    __shared__ alignas(16) unsigned short Psl[4 * 16 * 64];   // per-wave P tile

    const int tid  = threadIdx.x;
    const int lane = tid & 31;
    const int w    = tid >> 5;
    const int g    = lane >> 4;
    const int l15  = lane & 15;

    const int ntiles = Ndim / 64;               // 16
    const int bh = blockIdx.x / ntiles;
    const int qt = blockIdx.x % ntiles;
    const int b  = bh / Hh, h = bh % Hh;

    const size_t qbase  = (((size_t)(b * Hh + h)) * Ndim + qt * 64 + w * 16) * Dh;
    const size_t kvbase = ((size_t)(b * Hh + h)) * Mtot * Dh;

    // Q fragments (16 queries x 64, two k-steps of 32), loaded once
    Frag qf[2];
#pragma unroll
    for (int kk = 0; kk < 2; ++kk) {
        const unsigned short* qr = Q + qbase + (size_t)l15 * Dh + kk * 32 + (g << 4);
        qf[kk].q[0] = *(const uint4*)qr;
        qf[kk].q[1] = *(const uint4*)(qr + 8);
    }

    v8f zero = {0.f,0.f,0.f,0.f,0.f,0.f,0.f,0.f};
    v8f oacc[4];
#pragma unroll
    for (int i = 0; i < 4; ++i) oacc[i] = zero;
    float runm[8], lsum[8];
#pragma unroll
    for (int v = 0; v < 8; ++v) { runm[v] = -1e30f; lsum[v] = 0.f; }

    unsigned short* Pw = Psl + w * (16 * 64);
    const float scale = 0.125f;                 // d^-0.5, d=64

    const int nIter = Mtot / 64;                // 80
    for (int it = 0; it < nIter; ++it) {
        const size_t m0 = (size_t)it * 64;
        // ---- stage K tile row-major (64 keys x 64) via async b128 ----
        {
            int r = tid >> 1, half = tid & 1;
            const unsigned short* src = K + kvbase + (m0 + r) * Dh + half * 32;
            unsigned short* dst = Ksl + r * 64 + half * 32;
#pragma unroll
            for (int qd = 0; qd < 4; ++qd)
                copy16_g2l(src + (qd << 3), dst + (qd << 3));
        }
        // ---- stage V^T tile: Vtl[d][key], stride 80 (transpose in VGPRs) ----
#pragma unroll
        for (int rep = 0; rep < 4; ++rep) {
            int i = tid + (rep << 7);           // 0..511
            int r = i >> 3, c8 = i & 7;
            uint4 d = *(const uint4*)(V + kvbase + (m0 + r) * Dh + (c8 << 3));
            const unsigned short* e = (const unsigned short*)&d;
#pragma unroll
            for (int t2 = 0; t2 < 8; ++t2)
                Vtl[((c8 << 3) + t2) * 80 + r] = e[t2];
        }
        if (it + 1 < nIter)
            __builtin_prefetch(K + kvbase + (m0 + 64 + (tid >> 1)) * Dh, 0, 0);
        async_wait0();
        __syncthreads();

        // ---- scores S = Q K^T, 16x64 per wave ----
        v8f s[4];
#pragma unroll
        for (int nt = 0; nt < 4; ++nt) {
            v8f a = zero;
#pragma unroll
            for (int kk = 0; kk < 2; ++kk) {
                Frag bfr;
                const unsigned short* kr = Ksl + (nt * 16 + l15) * 64 + kk * 32 + (g << 4);
                bfr.q[0] = *(const uint4*)kr;
                bfr.q[1] = *(const uint4*)(kr + 8);
                a = __builtin_amdgcn_wmma_f32_16x16x32_bf16(
                    false, qf[kk].v, false, bfr.v, (short)0, a, false, false);
            }
            s[nt] = a;
        }

        // ---- online softmax: per-row (v,g) stats, reduce across 16 lanes ----
#pragma unroll
        for (int v = 0; v < 8; ++v) {
            float mx = -1e30f;
#pragma unroll
            for (int nt = 0; nt < 4; ++nt) { s[nt][v] *= scale; mx = fmaxf(mx, s[nt][v]); }
#pragma unroll
            for (int off = 1; off < 16; off <<= 1)
                mx = fmaxf(mx, __shfl_xor(mx, off, 32));
            float nm   = fmaxf(runm[v], mx);
            float corr = __expf(runm[v] - nm);
            runm[v] = nm;
            float ps = 0.f;
#pragma unroll
            for (int nt = 0; nt < 4; ++nt) {
                float pvl = __expf(s[nt][v] - nm);
                ps += pvl;
                Pw[(v + (g << 3)) * 64 + nt * 16 + l15] = f2bf(pvl);
            }
#pragma unroll
            for (int off = 1; off < 16; off <<= 1)
                ps += __shfl_xor(ps, off, 32);
            lsum[v] = lsum[v] * corr + ps;
#pragma unroll
            for (int nt = 0; nt < 4; ++nt) oacc[nt][v] *= corr;
        }
        asm volatile("s_wait_dscnt 0" ::: "memory");   // P writes -> frag reads

        // ---- O += P @ V : A-frag from Psl, B-frag from Vtl ----
#pragma unroll
        for (int kk = 0; kk < 2; ++kk) {
            Frag pf;
            const unsigned short* pr = Pw + (size_t)l15 * 64 + kk * 32 + (g << 4);
            pf.q[0] = *(const uint4*)pr;
            pf.q[1] = *(const uint4*)(pr + 8);
#pragma unroll
            for (int jt = 0; jt < 4; ++jt) {
                Frag vfr;
                const unsigned short* vr = Vtl + (jt * 16 + l15) * 80 + kk * 32 + (g << 4);
                vfr.q[0] = *(const uint4*)vr;
                vfr.q[1] = *(const uint4*)(vr + 8);
                oacc[jt] = __builtin_amdgcn_wmma_f32_16x16x32_bf16(
                    false, pf.v, false, vfr.v, (short)0, oacc[jt], false, false);
            }
        }
        __syncthreads();
    }

    // ---- normalize and write attention output [B,N,C] (token-major) ----
#pragma unroll
    for (int v = 0; v < 8; ++v) {
        float inv = 1.0f / lsum[v];
        int m = qt * 64 + w * 16 + v + (g << 3);
#pragma unroll
        for (int jt = 0; jt < 4; ++jt) {
            int c = h * Dh + jt * 16 + l15;
            O[((size_t)b * Ndim + m) * Cdim + c] = f2bf(oacc[jt][v] * inv);
        }
    }
}

// ---------------------------------------------------------------------------
// Orchestration
// ---------------------------------------------------------------------------
extern "C" void kernel_launch(void* const* d_in, const int* in_sizes, int n_in,
                              void* d_out, int out_size, void* d_ws, size_t ws_size,
                              hipStream_t stream) {
    const float* x_obj = (const float*)d_in[0];   // [2,1024,1024]
    const float* x_ctx = (const float*)d_in[1];   // [2,4096,1024]
    const float* Wq    = (const float*)d_in[2];   // [1024,1024]
    const float* Wkv   = (const float*)d_in[3];   // [1024,2048]
    const float* Wproj = (const float*)d_in[4];   // [1024,1024]
    const float* bproj = (const float*)d_in[5];   // [1024]

    const size_t SZ_XO = (size_t)Bb * Ndim * Cdim;       // 2,097,152
    const size_t SZ_XC = (size_t)Bb * Ldim * Cdim;       // 8,388,608
    const size_t SZ_WQ = (size_t)Cdim * Cdim;
    const size_t SZ_WKV = (size_t)Cdim * 2 * Cdim;
    const size_t SZ_WP = (size_t)Cdim * Cdim;
    const size_t SZ_Q  = (size_t)Bb * Hh * Ndim * Dh;
    const size_t SZ_KV = (size_t)Bb * Hh * Mtot * Dh;    // 10,485,760
    const size_t SZ_O  = (size_t)Bb * Ndim * Cdim;

    unsigned short* p   = (unsigned short*)d_ws;         // ~80 MB bf16 total
    unsigned short* xo  = p; p += SZ_XO;
    unsigned short* xc  = p; p += SZ_XC;
    unsigned short* wq  = p; p += SZ_WQ;
    unsigned short* wkv = p; p += SZ_WKV;
    unsigned short* wp  = p; p += SZ_WP;
    unsigned short* Qb  = p; p += SZ_Q;
    unsigned short* Kb  = p; p += SZ_KV;
    unsigned short* Vb  = p; p += SZ_KV;
    unsigned short* Ob  = p; p += SZ_O;

    cvt_f32_bf16<<<2048, 256, 0, stream>>>(x_obj, xo, (int)SZ_XO);
    cvt_f32_bf16<<<4096, 256, 0, stream>>>(x_ctx, xc, (int)SZ_XC);
    cvt_f32_bf16<<<1024, 256, 0, stream>>>(Wq,    wq, (int)SZ_WQ);
    cvt_f32_bf16<<<2048, 256, 0, stream>>>(Wkv,   wkv,(int)SZ_WKV);
    cvt_f32_bf16<<<1024, 256, 0, stream>>>(Wproj, wp, (int)SZ_WP);

    // Q = x_obj @ Wq   -> head-major Q_bf
    wmma_gemm_bf16<<<dim3(Cdim / 64, (Bb * Ndim) / 64), 128, 0, stream>>>(
        xo, wq, Cdim, Cdim, /*mode=*/0, Ndim, 0, Qb, nullptr, nullptr, nullptr);
    // kv_obj = x_obj @ Wkv -> K/V rows [0,N)
    wmma_gemm_bf16<<<dim3((2 * Cdim) / 64, (Bb * Ndim) / 64), 128, 0, stream>>>(
        xo, wkv, Cdim, 2 * Cdim, /*mode=*/1, Ndim, 0, Kb, Vb, nullptr, nullptr);
    // kv_ctx = x_ctx @ Wkv -> K/V rows [N, N+L)
    wmma_gemm_bf16<<<dim3((2 * Cdim) / 64, (Bb * Ldim) / 64), 128, 0, stream>>>(
        xc, wkv, Cdim, 2 * Cdim, /*mode=*/1, Ldim, Ndim, Kb, Vb, nullptr, nullptr);

    // fused attention: 2*16*16 = 512 blocks
    flash_attn_bf16<<<Bb * Hh * (Ndim / 64), 128, 0, stream>>>(Qb, Kb, Vb, Ob);

    // out = O @ Wproj + bproj  (f32 output)
    wmma_gemm_bf16<<<dim3(Cdim / 64, (Bb * Ndim) / 64), 128, 0, stream>>>(
        Ob, wp, Cdim, Cdim, /*mode=*/2, Bb * Ndim, 0, nullptr, nullptr,
        (float*)d_out, bproj);
}
